// NDNGeneration_83030307766793
// MI455X (gfx1250) — matrix operand order repaired and checked by reading, not compiled, not run on hardware
//
// CDNA5 / gfx1250 (MI455X) implementation of the layout-VAE forward pass.
// Core GEMMs use V_WMMA_F32_16X16X32_F16 (f16 in, fp32 accumulate, wave32).
// Key fusions / optimizations:
//   * gather/concat [obj[s] | pred | obj[o]] fused into WMMA A-operand loads,
//     fully unrolled via compile-time DIN (no branches in the K loop)
//   * B weight panel staged once per block in LDS (<=64KB of the 320KB WGP LDS),
//     read back as ds_load_b128 fragments; only A streams from global
//   * batched B-fragment loads: all NT fragments loaded into distinct registers
//     before the NT WMMAs, so ds waits retire with non-zero counts (ILP)
//   * software-pipelined A loads (prefetch k0+32 while issuing WMMAs for k0)
//   * scatter-add edge pooling fused into WMMA epilogue (global_atomic_add_f32)
//   * g_update (N,L,L,D) einsum collapsed algebraically to 2 row-variants +
//     prefix sums (h1 depends on (n,j) and only on the boolean k>j)
// Workspace requirement: ~380 MB (stage-B/C buffers alias the stage-A t1 buffer).
#include <hip/hip_runtime.h>
#include <hip/hip_bf16.h>
#include <cstdint>
#include <cstddef>

#define O_TOT 32768
#define P_TOT 131072
#define HIDN  512
#define DD    128
#define ZZ    32
#define NLAY  2048
#define LL    16
#define SLOPE 0.2f

typedef __attribute__((ext_vector_type(16))) _Float16 v16h;
typedef __attribute__((ext_vector_type(8)))  _Float16 v8h;
typedef __attribute__((ext_vector_type(8)))  float    v8f;

union AFrag { v16h v; v8h h[2]; };

__device__ __forceinline__ float lrelu(float x) { return x > 0.f ? x : SLOPE * x; }

// -------------------------------------------------------------------------
// Deterministic hash-based standard normal (stands in for jax.random eps).
__device__ __forceinline__ float gnoise(unsigned idx) {
    unsigned a = idx * 0x9E3779B9u + 0x7F4A7C15u;
    a ^= a >> 16; a *= 0x85EBCA6Bu; a ^= a >> 13; a *= 0xC2B2AE35u; a ^= a >> 16;
    unsigned b = idx * 0x632BE5ABu + 0x14057B7Fu;
    b ^= b >> 15; b *= 0x2C1B3C6Du; b ^= b >> 12; b *= 0x297A2D39u; b ^= b >> 15;
    float u1 = ((a >> 8) + 1u) * (1.0f / 16777217.0f);   // (0,1]
    float u2 = (b >> 8) * (1.0f / 16777216.0f);          // [0,1)
    return sqrtf(-2.0f * logf(u1)) * cosf(6.28318530718f * u2);
}

// -------------------------------------------------------------------------
// WMMA GEMM:  C[M,N] = act( A[M,K](f16) @ Wt(f16, col-major [n*K+k]) + bias )
// 256 threads = 8 wave32s; each wave computes a 16 x (16*NT) tile; all 8 waves
// share one LDS-resident B panel of NT*16 columns x K (contiguous in Wt).
// A modes : direct row-major (lda), or 3-segment gather [obj[s]|pred|obj[o]]
//           with compile-time segment width DIN (fully unrolled K loop).
// Epilogue: store f16 / store f32 / scatter (cols<512 -> atomicAdd pooled[s],
//           512..639 -> pv f16, >=640 -> atomicAdd pooled[o]).
// NOTE: launcher guarantees M % 128 == 0, N % (16*NT) == 0, K % 32 == 0.
constexpr int A_DIRECT = 0, A_GATHER3 = 1;
constexpr int E_F16 = 0, E_F32 = 1, E_SCATTER = 2;

__device__ __forceinline__ AFrag load_afrag(const _Float16* __restrict__ base, int ko, int g) {
    // A 16x32 f16 fragment: m = lane%16; halves i<8 -> k=ko+g*8+i, i>=8 -> k=ko+16+g*8+(i-8)
    AFrag f;
    f.h[0] = *(const v8h*)(base + ko + g * 8);
    f.h[1] = *(const v8h*)(base + ko + 16 + g * 8);
    return f;
}

template<int AM, int EM, int NT, int DIN>
__global__ __launch_bounds__(256) void wmma_gemm(
    const _Float16* __restrict__ A, int lda,
    const _Float16* __restrict__ objF,
    const _Float16* __restrict__ predF,
    const int* __restrict__ sidx, const int* __restrict__ oidx,
    const _Float16* __restrict__ Wt,
    const float* __restrict__ bias,
    int M, int N, int K, int act,
    _Float16* __restrict__ outH, float* __restrict__ outF, int ldo,
    float* __restrict__ pooled, _Float16* __restrict__ pvOut)
{
    extern __shared__ _Float16 ldsB[];   // NT*16 columns x K halves (col-major)

    const int lane = threadIdx.x & 31;
    const int wave = threadIdx.x >> 5;
    const int m0 = (blockIdx.y * 8 + wave) * 16;
    const int n0 = blockIdx.x * (16 * NT);
    const int lm = lane & 15;            // A row / B,C column within tile
    const int g  = lane >> 4;            // half-wave group
    const int row = m0 + lm;

    // ---- cooperative B-panel load: Wt[n0*K .. (n0+16*NT)*K) is contiguous
    {
        const v8h* __restrict__ src = (const v8h*)(Wt + (size_t)n0 * K);
        v8h* __restrict__ dst = (v8h*)ldsB;
        const int nch = (NT * 16 * K) >> 3;
        for (int c = threadIdx.x; c < nch; c += 256) dst[c] = src[c];
    }
    __syncthreads();

    v8f acc[NT];
#pragma unroll
    for (int t = 0; t < NT; ++t)
#pragma unroll
        for (int r = 0; r < 8; ++r) acc[t][r] = 0.f;

    // B fragment from LDS: n = lane%16; i<8 -> k=kg+g*8+i, i>=8 -> k=kg+16+g*8+(i-8)
    auto bfrag = [&](int t, int kg) {
        const _Float16* bp = ldsB + (size_t)(t * 16 + lm) * K + kg + g * 8;
        AFrag b;
        b.h[0] = *(const v8h*)(bp);
        b.h[1] = *(const v8h*)(bp + 16);
        return b;
    };
    // load-all-then-compute so the scheduler batches ds_loads across tiles
    auto kstep = [&](const AFrag& a_cur, int kg) {
        AFrag b[NT];
#pragma unroll
        for (int t = 0; t < NT; ++t) b[t] = bfrag(t, kg);
#pragma unroll
        for (int t = 0; t < NT; ++t)
            acc[t] = __builtin_amdgcn_wmma_f32_16x16x32_f16(
                false, a_cur.v, false, b[t].v, (short)0, acc[t], false, false);
    };

    if (AM == A_GATHER3) {
        // ---- fully unrolled 3-segment gather GEMM (K == 3*DIN)
        const _Float16* bases[3];
        bases[0] = objF + (size_t)sidx[row] * DIN;
        bases[1] = predF + (size_t)row * DIN;
        bases[2] = objF + (size_t)oidx[row] * DIN;
        AFrag a = load_afrag(bases[0], 0, g);
        int kg = 0;
#pragma unroll
        for (int seg = 0; seg < 3; ++seg) {
#pragma unroll
            for (int ko = 0; ko < DIN; ko += 32) {
                AFrag a_cur = a;
                int nseg = seg, nko = ko + 32;          // software pipeline: prefetch next
                if (nko >= DIN) { nseg = seg + 1; nko = 0; }
                if (nseg < 3) a = load_afrag(bases[nseg], nko, g);
                kstep(a_cur, kg);
                kg += 32;
            }
        }
    } else {
        const _Float16* aRow = A + (size_t)row * lda;
        AFrag a = load_afrag(aRow, 0, g);
        for (int k0 = 0; k0 < K; k0 += 32) {
            AFrag a_cur = a;
            if (k0 + 32 < K) {                          // software pipeline
                a = load_afrag(aRow, k0 + 32, g);
                __builtin_prefetch(aRow + k0 + 96, 0, 1);
            }
            kstep(a_cur, k0);
        }
    }

    // ---- epilogue: C/D layout m = r + 8*g, n = lane%16
#pragma unroll
    for (int t = 0; t < NT; ++t) {
        const int nn = n0 + t * 16 + lm;
        const float bv = bias ? bias[nn] : 0.f;
#pragma unroll
        for (int r = 0; r < 8; ++r) {
            const int mm = m0 + r + 8 * g;
            float v = acc[t][r] + bv;
            if (act) v = lrelu(v);
            if (EM == E_F16) {
                outH[(size_t)mm * ldo + nn] = (_Float16)v;
            } else if (EM == E_F32) {
                outF[(size_t)mm * ldo + nn] = v;
            } else { // E_SCATTER: t[:, :512]->pooled[s] ; [512:640]->pv ; [640:]->pooled[o]
                if (nn < HIDN)
                    atomicAdd(pooled + (size_t)sidx[mm] * HIDN + nn, v);
                else if (nn < HIDN + DD)
                    pvOut[(size_t)mm * DD + (nn - HIDN)] = (_Float16)v;
                else
                    atomicAdd(pooled + (size_t)oidx[mm] * HIDN + (nn - HIDN - DD), v);
            }
        }
    }
}

// -------------------------------------------------------------------------
// Small helper kernels
__global__ void k_f2h(const float* __restrict__ s, _Float16* __restrict__ d, size_t n) {
    size_t i = (size_t)blockIdx.x * blockDim.x + threadIdx.x;
    if (i < n) d[i] = (_Float16)s[i];
}

// w[K x N] row-major f32 -> wt[N x K] column-major f16 (WMMA B-operand layout)
__global__ void k_w_cm(const float* __restrict__ w, _Float16* __restrict__ wt, int K, int N) {
    int i = blockIdx.x * blockDim.x + threadIdx.x;
    if (i >= K * N) return;
    int n = i % N, k = i / N;
    wt[(size_t)n * K + k] = (_Float16)w[i];
}

__global__ void k_count(const int* __restrict__ s, const int* __restrict__ o,
                        float* __restrict__ cnt) {
    int i = blockIdx.x * blockDim.x + threadIdx.x;
    if (i >= P_TOT) return;
    atomicAdd(cnt + s[i], 1.f);
    atomicAdd(cnt + o[i], 1.f);
}

__global__ void k_pool_mean(const float* __restrict__ pooled, const float* __restrict__ cnt,
                            _Float16* __restrict__ pm) {
    size_t i = (size_t)blockIdx.x * blockDim.x + threadIdx.x;
    if (i >= (size_t)O_TOT * HIDN) return;
    float c = cnt[i / HIDN];
    c = c > 1.f ? c : 1.f;
    pm[i] = (_Float16)(pooled[i] / c);
}

// A/B row variants of g_update h1:  A=lrelu(fproj+bxproj+b1), B=lrelu(fproj+b1)
__global__ void k_bx_ab(const float* __restrict__ fproj, const float* __restrict__ boxes,
                        const float* __restrict__ w1, const float* __restrict__ b1,
                        _Float16* __restrict__ ab) {
    size_t i = (size_t)blockIdx.x * blockDim.x + threadIdx.x;
    if (i >= (size_t)O_TOT * DD) return;
    int o = (int)(i / DD), h = (int)(i % DD);
    float fp = fproj[i] + b1[h];
    float bp = 0.f;
#pragma unroll
    for (int d = 0; d < 4; ++d) bp += boxes[o * 4 + d] * w1[(size_t)(DD + d) * DD + h];
    ab[i] = (_Float16)lrelu(fp + bp);
    ab[(size_t)O_TOT * DD + i] = (_Float16)lrelu(fp);
}

// c[n,k] = ( sum_{j<k} hA[n,j] + sum_{j>=k} hB[n,j] ) / 16
__global__ void k_prefix_c(const float* __restrict__ hab, _Float16* __restrict__ cf16) {
    int n = blockIdx.x, h = threadIdx.x;            // 2048 blocks x 128 threads
    const float* hA = hab + (size_t)(n * LL) * DD + h;
    const float* hB = hab + (size_t)(O_TOT + n * LL) * DD + h;
    float sumB = 0.f;
    for (int j = 0; j < LL; ++j) sumB += hB[(size_t)j * DD];
    float run = 0.f;
    for (int k = 0; k < LL; ++k) {
        float cv = (run + sumB) * (1.f / 16.f);
        cf16[(size_t)(n * LL + k) * DD + h] = (_Float16)cv;
        run += hA[(size_t)k * DD] - hB[(size_t)k * DD];
    }
}

// enc.bb (K=4) into columns [0,128) of the 256-wide concat buffer
__global__ void k_bb(const float* __restrict__ boxes, const float* __restrict__ w,
                     const float* __restrict__ b, _Float16* __restrict__ cat) {
    size_t i = (size_t)blockIdx.x * blockDim.x + threadIdx.x;
    if (i >= (size_t)O_TOT * DD) return;
    int o = (int)(i / DD), h = (int)(i % DD);
    float v = b[h];
#pragma unroll
    for (int d = 0; d < 4; ++d) v += boxes[o * 4 + d] * w[d * DD + h];
    cat[(size_t)o * 256 + h] = (_Float16)lrelu(v);
}

// z = eps*exp(0.5*var)+mu into cols [0,32), c into cols [32,160) of dec concat
__global__ void k_z_cat(const float* __restrict__ mu, const float* __restrict__ var,
                        const _Float16* __restrict__ cf16, _Float16* __restrict__ zc) {
    size_t i = (size_t)blockIdx.x * blockDim.x + threadIdx.x;
    if (i >= (size_t)O_TOT * 160) return;
    int o = (int)(i / 160), j = (int)(i % 160);
    if (j < ZZ) {
        float m = mu[(size_t)o * ZZ + j], v = var[(size_t)o * ZZ + j];
        zc[i] = (_Float16)(gnoise((unsigned)(o * ZZ + j)) * expf(0.5f * v) + m);
    } else {
        zc[i] = cf16[(size_t)o * DD + (j - ZZ)];
    }
}

// dec.l3: (O x 64) @ (64 x 4) + b, no activation -> pred_boxes (f32)
__global__ void k_dec3(const _Float16* __restrict__ x, const float* __restrict__ w,
                       const float* __restrict__ b, float* __restrict__ out) {
    size_t i = (size_t)blockIdx.x * blockDim.x + threadIdx.x;
    if (i >= (size_t)O_TOT * 4) return;
    int o = (int)(i / 4), j = (int)(i % 4);
    float v = b[j];
    for (int k = 0; k < 64; ++k) v += (float)x[(size_t)o * 64 + k] * w[k * 4 + j];
    out[i] = v;
}

// -------------------------------------------------------------------------
// Input index map: params flattened in setup_inputs() insertion order
// (_mk_dense inserts 'w' then 'b'), then objs, obj_vecs, pred_vecs, boxes, s_idx, o_idx.
enum {
    IDX_GENC = 0,            // 3 layers x {n1a.w,n1a.b,n1b.w,n1b.b,n2a.w,n2a.b,n2b.w,n2b.b}
    IDX_GU_L1W = 24, IDX_GU_L1B, IDX_GU_L2W, IDX_GU_L2B,
    IDX_ENC_BBW = 28, IDX_ENC_BBB, IDX_ENC_CW, IDX_ENC_CB, IDX_ENC_HW, IDX_ENC_HB,
    IDX_ENC_MUW, IDX_ENC_MUB, IDX_ENC_VARW, IDX_ENC_VARB,
    IDX_DEC_L1W = 38, IDX_DEC_L1B, IDX_DEC_L2W, IDX_DEC_L2B, IDX_DEC_L3W, IDX_DEC_L3B,
    IDX_PR_L1W = 44, IDX_PR_L1B, IDX_PR_L2W, IDX_PR_L2B, IDX_PR_L3W, IDX_PR_L3B,
    IDX_PR_MUW, IDX_PR_MUB, IDX_PR_VARW, IDX_PR_VARB,
    IDX_OBJS = 54, IDX_OBJ_VECS, IDX_PRED_VECS, IDX_BOXES, IDX_SIDX, IDX_OIDX
};

extern "C" void kernel_launch(void* const* d_in, const int* in_sizes, int n_in,
                              void* d_out, int out_size, void* d_ws, size_t ws_size,
                              hipStream_t stream) {
    (void)in_sizes; (void)n_in; (void)out_size; (void)ws_size;
    auto F = [&](int i) { return (const float*)d_in[i]; };
    const int* sidx = (const int*)d_in[IDX_SIDX];
    const int* oidx = (const int*)d_in[IDX_OIDX];
    const float* obj_vecs = F(IDX_OBJ_VECS);
    const float* pred_vecs = F(IDX_PRED_VECS);
    const float* boxes = F(IDX_BOXES);

    // ---- workspace bump allocator (~380 MB total; stage-B/C aliases t1)
    uint8_t* base = (uint8_t*)d_ws;
    size_t off = 0;
    auto alloc = [&](size_t bytes) -> void* {
        off = (off + 255) & ~(size_t)255;
        void* p = base + off; off += bytes; return p;
    };
    _Float16* obj64   = (_Float16*)alloc((size_t)O_TOT * 64 * 2);
    _Float16* pred64  = (_Float16*)alloc((size_t)P_TOT * 64 * 2);
    _Float16* obj128a = (_Float16*)alloc((size_t)O_TOT * DD * 2);
    _Float16* obj128b = (_Float16*)alloc((size_t)O_TOT * DD * 2);
    _Float16* pv128a  = (_Float16*)alloc((size_t)P_TOT * DD * 2);
    _Float16* pv128b  = (_Float16*)alloc((size_t)P_TOT * DD * 2);
    _Float16* t1      = (_Float16*)alloc((size_t)P_TOT * HIDN * 2);   // 134 MB
    float*    pooled  = (float*)   alloc((size_t)O_TOT * HIDN * 4);   // 67 MB
    float*    cnt     = (float*)   alloc((size_t)O_TOT * 4);
    _Float16* pm      = (_Float16*)alloc((size_t)O_TOT * HIDN * 2);
    _Float16* t2      = (_Float16*)alloc((size_t)O_TOT * HIDN * 2);
    _Float16* wt      = (_Float16*)alloc((size_t)HIDN * 1152 * 2);    // weight slot (reused)

    // stage-B/C buffers alias t1 (dead after stage A) and pooled/pm.
    uint8_t* s2 = (uint8_t*)t1; size_t off2 = 0;
    auto alloc2 = [&](size_t bytes) -> void* {
        off2 = (off2 + 255) & ~(size_t)255;
        void* p = s2 + off2; off2 += bytes; return p;
    };
    float* fproj = (float*)pooled;                 // alias: pooled dead after stage A
    _Float16* ab = (_Float16*)pm;                  // alias: pm dead after stage A
    float*    hab    = (float*)   alloc2((size_t)2 * O_TOT * DD * 4);
    _Float16* cf16   = (_Float16*)alloc2((size_t)O_TOT * DD * 2);
    _Float16* cat256 = (_Float16*)alloc2((size_t)O_TOT * 256 * 2);
    _Float16* henc   = (_Float16*)alloc2((size_t)O_TOT * ZZ * 2);
    _Float16* zc     = (_Float16*)alloc2((size_t)O_TOT * 160 * 2);
    _Float16* dl1    = (_Float16*)alloc2((size_t)O_TOT * DD * 2);
    _Float16* dl2    = (_Float16*)alloc2((size_t)O_TOT * 64 * 2);
    _Float16* pr1    = (_Float16*)alloc2((size_t)O_TOT * DD * 2);
    _Float16* pr2    = (_Float16*)alloc2((size_t)O_TOT * DD * 2);
    _Float16* pr3    = (_Float16*)alloc2((size_t)O_TOT * ZZ * 2);

    float* outPB   = (float*)d_out;                         // (2048,16,4)
    float* outMu   = outPB + (size_t)O_TOT * 4;             // (2048,16,32)
    float* outVar  = outMu + (size_t)O_TOT * ZZ;
    float* outMuP  = outVar + (size_t)O_TOT * ZZ;
    float* outVarP = outMuP + (size_t)O_TOT * ZZ;

    const dim3 blk(256);
    auto gg = [](int M, int N, int NT) { return dim3((unsigned)(N / (16 * NT)), (unsigned)(M / 128)); };
    auto eg = [](size_t n) { return dim3((unsigned)((n + 255) / 256)); };
    auto shm = [](int NT, int K) { return (size_t)NT * 16 * K * 2; };  // LDS B panel bytes

    // ---- prologue: f16 features, edge counts
    k_f2h<<<eg((size_t)O_TOT * 64), blk, 0, stream>>>(obj_vecs, obj64, (size_t)O_TOT * 64);
    k_f2h<<<eg((size_t)P_TOT * 64), blk, 0, stream>>>(pred_vecs, pred64, (size_t)P_TOT * 64);
    hipMemsetAsync(cnt, 0, (size_t)O_TOT * 4, stream);
    k_count<<<eg(P_TOT), blk, 0, stream>>>(sidx, oidx, cnt);

    // ---- stage A: 3 graph-conv layers
    const _Float16* objCur = obj64;
    const _Float16* predCur = pred64;
    _Float16* objBuf[2] = { obj128a, obj128b };
    _Float16* pvBuf[2]  = { pv128a, pv128b };
    for (int l = 0; l < 3; ++l) {
        const float* w1a = F(l * 8 + 0); const float* b1a = F(l * 8 + 1);
        const float* w1b = F(l * 8 + 2); const float* b1b = F(l * 8 + 3);
        const float* w2a = F(l * 8 + 4); const float* b2a = F(l * 8 + 5);
        const float* w2b = F(l * 8 + 6); const float* b2b = F(l * 8 + 7);
        const int din = (l == 0) ? 64 : 128;
        const int K1 = 3 * din;

        k_w_cm<<<eg((size_t)K1 * HIDN), blk, 0, stream>>>(w1a, wt, K1, HIDN);
        if (l == 0)
            wmma_gemm<A_GATHER3, E_F16, 4, 64><<<gg(P_TOT, HIDN, 4), blk, shm(4, K1), stream>>>(
                nullptr, 0, objCur, predCur, sidx, oidx, wt, b1a,
                P_TOT, HIDN, K1, 1, t1, nullptr, HIDN, nullptr, nullptr);
        else
            wmma_gemm<A_GATHER3, E_F16, 4, 128><<<gg(P_TOT, HIDN, 4), blk, shm(4, K1), stream>>>(
                nullptr, 0, objCur, predCur, sidx, oidx, wt, b1a,
                P_TOT, HIDN, K1, 1, t1, nullptr, HIDN, nullptr, nullptr);

        hipMemsetAsync(pooled, 0, (size_t)O_TOT * HIDN * 4, stream);
        k_w_cm<<<eg((size_t)HIDN * 1152), blk, 0, stream>>>(w1b, wt, HIDN, 1152);
        wmma_gemm<A_DIRECT, E_SCATTER, 4, 0><<<gg(P_TOT, 1152, 4), blk, shm(4, HIDN), stream>>>(
            t1, HIDN, nullptr, nullptr, sidx, oidx, wt, b1b,
            P_TOT, 1152, HIDN, 1, nullptr, nullptr, 0, pooled, pvBuf[l & 1]);

        k_pool_mean<<<eg((size_t)O_TOT * HIDN), blk, 0, stream>>>(pooled, cnt, pm);
        k_w_cm<<<eg((size_t)HIDN * HIDN), blk, 0, stream>>>(w2a, wt, HIDN, HIDN);
        wmma_gemm<A_DIRECT, E_F16, 4, 0><<<gg(O_TOT, HIDN, 4), blk, shm(4, HIDN), stream>>>(
            pm, HIDN, nullptr, nullptr, nullptr, nullptr, wt, b2a,
            O_TOT, HIDN, HIDN, 1, t2, nullptr, HIDN, nullptr, nullptr);
        k_w_cm<<<eg((size_t)HIDN * DD), blk, 0, stream>>>(w2b, wt, HIDN, DD);
        wmma_gemm<A_DIRECT, E_F16, 4, 0><<<gg(O_TOT, DD, 4), blk, shm(4, HIDN), stream>>>(
            t2, HIDN, nullptr, nullptr, nullptr, nullptr, wt, b2b,
            O_TOT, DD, HIDN, 1, objBuf[l & 1], nullptr, DD, nullptr, nullptr);
        objCur = objBuf[l & 1];
        predCur = pvBuf[l & 1];
    }

    // ---- stage B: g_update, collapsed to 2 row variants + prefix sums
    const float* guW1 = F(IDX_GU_L1W);   // (132,128) row-major; rows 0..127 used for fproj
    k_w_cm<<<eg((size_t)DD * DD), blk, 0, stream>>>(guW1, wt, DD, DD);
    wmma_gemm<A_DIRECT, E_F32, 4, 0><<<gg(O_TOT, DD, 4), blk, shm(4, DD), stream>>>(
        objCur, DD, nullptr, nullptr, nullptr, nullptr, wt, nullptr,
        O_TOT, DD, DD, 0, nullptr, fproj, DD, nullptr, nullptr);
    k_bx_ab<<<eg((size_t)O_TOT * DD), blk, 0, stream>>>(fproj, boxes, guW1, F(IDX_GU_L1B), ab);
    k_w_cm<<<eg((size_t)DD * DD), blk, 0, stream>>>(F(IDX_GU_L2W), wt, DD, DD);
    wmma_gemm<A_DIRECT, E_F32, 4, 0><<<gg(2 * O_TOT, DD, 4), blk, shm(4, DD), stream>>>(
        ab, DD, nullptr, nullptr, nullptr, nullptr, wt, F(IDX_GU_L2B),
        2 * O_TOT, DD, DD, 1, nullptr, hab, DD, nullptr, nullptr);
    k_prefix_c<<<dim3(NLAY), dim3(DD), 0, stream>>>(hab, cf16);

    // ---- stage C: encoder / z / decoder / prior
    k_bb<<<eg((size_t)O_TOT * DD), blk, 0, stream>>>(boxes, F(IDX_ENC_BBW), F(IDX_ENC_BBB), cat256);
    k_w_cm<<<eg((size_t)DD * DD), blk, 0, stream>>>(F(IDX_ENC_CW), wt, DD, DD);
    wmma_gemm<A_DIRECT, E_F16, 4, 0><<<gg(O_TOT, DD, 4), blk, shm(4, DD), stream>>>(
        cf16, DD, nullptr, nullptr, nullptr, nullptr, wt, F(IDX_ENC_CB),
        O_TOT, DD, DD, 1, cat256 + DD, nullptr, 256, nullptr, nullptr);
    k_w_cm<<<eg((size_t)256 * ZZ), blk, 0, stream>>>(F(IDX_ENC_HW), wt, 256, ZZ);
    wmma_gemm<A_DIRECT, E_F16, 2, 0><<<gg(O_TOT, ZZ, 2), blk, shm(2, 256), stream>>>(
        cat256, 256, nullptr, nullptr, nullptr, nullptr, wt, F(IDX_ENC_HB),
        O_TOT, ZZ, 256, 1, henc, nullptr, ZZ, nullptr, nullptr);
    k_w_cm<<<eg((size_t)ZZ * ZZ), blk, 0, stream>>>(F(IDX_ENC_MUW), wt, ZZ, ZZ);
    wmma_gemm<A_DIRECT, E_F32, 2, 0><<<gg(O_TOT, ZZ, 2), blk, shm(2, ZZ), stream>>>(
        henc, ZZ, nullptr, nullptr, nullptr, nullptr, wt, F(IDX_ENC_MUB),
        O_TOT, ZZ, ZZ, 1, nullptr, outMu, ZZ, nullptr, nullptr);
    k_w_cm<<<eg((size_t)ZZ * ZZ), blk, 0, stream>>>(F(IDX_ENC_VARW), wt, ZZ, ZZ);
    wmma_gemm<A_DIRECT, E_F32, 2, 0><<<gg(O_TOT, ZZ, 2), blk, shm(2, ZZ), stream>>>(
        henc, ZZ, nullptr, nullptr, nullptr, nullptr, wt, F(IDX_ENC_VARB),
        O_TOT, ZZ, ZZ, 1, nullptr, outVar, ZZ, nullptr, nullptr);

    k_z_cat<<<eg((size_t)O_TOT * 160), blk, 0, stream>>>(outMu, outVar, cf16, zc);
    k_w_cm<<<eg((size_t)160 * DD), blk, 0, stream>>>(F(IDX_DEC_L1W), wt, 160, DD);
    wmma_gemm<A_DIRECT, E_F16, 4, 0><<<gg(O_TOT, DD, 4), blk, shm(4, 160), stream>>>(
        zc, 160, nullptr, nullptr, nullptr, nullptr, wt, F(IDX_DEC_L1B),
        O_TOT, DD, 160, 1, dl1, nullptr, DD, nullptr, nullptr);
    k_w_cm<<<eg((size_t)DD * 64), blk, 0, stream>>>(F(IDX_DEC_L2W), wt, DD, 64);
    wmma_gemm<A_DIRECT, E_F16, 4, 0><<<gg(O_TOT, 64, 4), blk, shm(4, DD), stream>>>(
        dl1, DD, nullptr, nullptr, nullptr, nullptr, wt, F(IDX_DEC_L2B),
        O_TOT, 64, DD, 1, dl2, nullptr, 64, nullptr, nullptr);
    k_dec3<<<eg((size_t)O_TOT * 4), blk, 0, stream>>>(dl2, F(IDX_DEC_L3W), F(IDX_DEC_L3B), outPB);

    k_w_cm<<<eg((size_t)DD * DD), blk, 0, stream>>>(F(IDX_PR_L1W), wt, DD, DD);
    wmma_gemm<A_DIRECT, E_F16, 4, 0><<<gg(O_TOT, DD, 4), blk, shm(4, DD), stream>>>(
        cf16, DD, nullptr, nullptr, nullptr, nullptr, wt, F(IDX_PR_L1B),
        O_TOT, DD, DD, 1, pr1, nullptr, DD, nullptr, nullptr);
    k_w_cm<<<eg((size_t)DD * DD), blk, 0, stream>>>(F(IDX_PR_L2W), wt, DD, DD);
    wmma_gemm<A_DIRECT, E_F16, 4, 0><<<gg(O_TOT, DD, 4), blk, shm(4, DD), stream>>>(
        pr1, DD, nullptr, nullptr, nullptr, nullptr, wt, F(IDX_PR_L2B),
        O_TOT, DD, DD, 1, pr2, nullptr, DD, nullptr, nullptr);
    k_w_cm<<<eg((size_t)DD * ZZ), blk, 0, stream>>>(F(IDX_PR_L3W), wt, DD, ZZ);
    wmma_gemm<A_DIRECT, E_F16, 2, 0><<<gg(O_TOT, ZZ, 2), blk, shm(2, DD), stream>>>(
        pr2, DD, nullptr, nullptr, nullptr, nullptr, wt, F(IDX_PR_L3B),
        O_TOT, ZZ, DD, 1, pr3, nullptr, ZZ, nullptr, nullptr);
    k_w_cm<<<eg((size_t)ZZ * ZZ), blk, 0, stream>>>(F(IDX_PR_MUW), wt, ZZ, ZZ);
    wmma_gemm<A_DIRECT, E_F32, 2, 0><<<gg(O_TOT, ZZ, 2), blk, shm(2, ZZ), stream>>>(
        pr3, ZZ, nullptr, nullptr, nullptr, nullptr, wt, F(IDX_PR_MUB),
        O_TOT, ZZ, ZZ, 1, nullptr, outMuP, ZZ, nullptr, nullptr);
    k_w_cm<<<eg((size_t)ZZ * ZZ), blk, 0, stream>>>(F(IDX_PR_VARW), wt, ZZ, ZZ);
    wmma_gemm<A_DIRECT, E_F32, 2, 0><<<gg(O_TOT, ZZ, 2), blk, shm(2, ZZ), stream>>>(
        pr3, ZZ, nullptr, nullptr, nullptr, nullptr, wt, F(IDX_PR_VARB),
        O_TOT, ZZ, ZZ, 1, nullptr, outVarP, ZZ, nullptr, nullptr);
}
